// PoseCDE_5987184411237
// MI455X (gfx1250) — compile-verified, matrix operand order checked by reading, not run on hardware
//
// PoseCDE on MI455X (gfx1250, wave32, WMMA).
//
// Algebraic collapse (holds for ANY inputs, by reference semantics):
//   dXdt(t) = derivs[:, 0] for every t the RK4 sweep touches (ceil(t)-1 == 0
//   on the fixed unit knot grid), and derivs[:,0] = (ts[:,2]-ts[:,1], 0,...,0)
//   since the rectilinear value channels cancel. Hence
//     f(z)[b,h] = dt_b * tanh( (relu(relu(z@Wf0+bf0)@Wf1+bf1) @ Wout[:,h*C]) + bout[h*C] )
//   and fv/fi provably do not influence the output: the 262656-col GEMM
//   collapses to a 512-col GEMM with Wsel = Wout[:, 0::513].
//
// Workload: 108 dependent [64x512]x[512x512] GEMMs via v_wmma_f32_16x16x32_bf16.
// All GEMM operands (weights AND activations) live as pre-packed bf16 per-lane
// fragment images, so the inner loop is: 2x contiguous 32B loads + prefetch +
// WMMA, with no fp32->bf16 VALU conversion on the critical path (the round-1
// version burned ~48 VALU per WMMA on redundant per-N-tile conversions).

#include <hip/hip_runtime.h>
#include <hip/hip_bf16.h>
#include <math.h>

typedef __attribute__((ext_vector_type(16))) __bf16 v16bf;
typedef __attribute__((ext_vector_type(8)))  float  v8f;

#define B_    64
#define Lp1_  64      // L+1
#define H_    512
#define C_    513     // H+1
#define KD_   512
#define ND_   512
#define NEV_  10

// float -> bf16, round-to-nearest-even, pure bit ops
__device__ __forceinline__ __bf16 f2bf(float f) {
  unsigned u = __builtin_bit_cast(unsigned, f);
  u += 0x7FFFu + ((u >> 16) & 1u);
  unsigned short s = (unsigned short)(u >> 16);
  return __builtin_bit_cast(__bf16, s);
}

// Fragment-image flat index for matrix element (m, k) of a [64 x 512] operand
// laid out as 16-bit 16x32 WMMA fragments (ISA 7.12.2):
//   tile mt = m>>4, k-step s = k>>5, rr = k&31,
//   lane = (m&15) + 16*((rr>>3)&1),  e = (rr&7) + 8*(rr>>4)
//   idx  = ((mt*16 + s)*32 + lane)*16 + e
__device__ __forceinline__ long frag_idx(int m, int k) {
  int mt = m >> 4, s = k >> 5, rr = k & 31;
  int lane = (m & 15) + 16 * ((rr >> 3) & 1);
  int e = (rr & 7) + 8 * (rr >> 4);
  return ((((long)mt * 16 + s) * 32 + lane) << 4) + e;
}

// ---------------------------------------------------------------------------
// Pack W[k][n] (element at src[k*rs + n*cs]) into a bf16 B-fragment image.
// ---------------------------------------------------------------------------
__global__ __launch_bounds__(256)
void pack_w_kernel(const float* __restrict__ src, long rs, long cs,
                   __bf16* __restrict__ dst) {
  int i = blockIdx.x * blockDim.x + threadIdx.x;
  if (i >= H_ * KD_) return;
  int e    = i & 15;
  int lane = (i >> 4) & 31;
  int s    = (i >> 9) & 15;
  int nt   = i >> 13;
  int n = nt * 16 + (lane & 15);
  int k = 32 * s + (e & 7) + 16 * (e >> 3) + 8 * (lane >> 4);
  dst[i] = f2bf(src[(long)k * rs + (long)n * cs]);
}

// bsel[h] = bout[h*C]; dt[b] = ts[b,2]-ts[b,1]; z = 0 (fp32 + frag image)
__global__ __launch_bounds__(256)
void prep_misc_kernel(const float* __restrict__ ts, const float* __restrict__ bout,
                      float* __restrict__ bsel, float* __restrict__ dt,
                      float* __restrict__ z, __bf16* __restrict__ zfrag) {
  int i = blockIdx.x * blockDim.x + threadIdx.x;
  if (i < H_)      bsel[i] = bout[(long)i * C_];
  if (i < B_)      dt[i]   = ts[i * Lp1_ + 2] - ts[i * Lp1_ + 1];
  if (i < B_ * H_) { z[i] = 0.0f; zfrag[i] = f2bf(0.0f); }
}

// ---------------------------------------------------------------------------
// GEMM: [64x512] @ [512x512] + bias, operands as bf16 fragment images.
// One wave per 16x16 output tile; grid = (4 Mtiles, 32 Ntiles).
// mode 0: relu(x+bias)  -> written as bf16 A-fragment image for the next GEMM
// mode 1: dt[b]*tanh(x+bias) -> written as plain fp32 rows (elementwise use)
// EXEC all-ones for the whole wave (WMMA requirement).
// ---------------------------------------------------------------------------
__global__ __launch_bounds__(32)
void gemm_wmma_kernel(const __bf16* __restrict__ Ap,   // A fragment image
                      const __bf16* __restrict__ Wp,   // B fragment image
                      const float* __restrict__ bias,
                      const float* __restrict__ dt,
                      __bf16* __restrict__ outFrag,    // mode 0 destination
                      float* __restrict__ outF32,      // mode 1 destination
                      int mode) {
  const int lane = threadIdx.x;          // 0..31
  const int mt   = blockIdx.x;           // 0..3
  const int nt   = blockIdx.y;           // 0..31
  const int ml   = lane & 15;
  const int bh   = lane >> 4;

  const __bf16* __restrict__ apb = Ap + (((long)mt * 16 * 32 + lane) << 4);
  const __bf16* __restrict__ bpb = Wp + (((long)nt * 16 * 32 + lane) << 4);

  v8f acc = {};
#pragma unroll 4
  for (int s = 0; s < 16; ++s) {
    v16bf af = *(const v16bf*)(apb + ((long)s * 32 << 4));
    v16bf bf = *(const v16bf*)(bpb + ((long)s * 32 << 4));
    if (s + 1 < 16) {   // warm next fragment lines: global_prefetch_b8
      __builtin_prefetch(apb + (((long)s + 1) * 32 << 4), 0, 1);
      __builtin_prefetch(bpb + (((long)s + 1) * 32 << 4), 0, 1);
    }
    acc = __builtin_amdgcn_wmma_f32_16x16x32_bf16(
        /*neg_a=*/false, af, /*neg_b=*/false, bf,
        /*c_mod=*/(short)0, acc, /*reuse_a=*/false, /*reuse_b=*/false);
  }

  // C/D layout: vgpr r -> row (mt*16 + r + 8*bh); col n = nt*16 + (lane&15)
  const int n  = nt * 16 + ml;
  const float bn = bias[n];
  if (mode == 0) {
    // Re-emit as A-fragment image of the NEXT gemm: element (m, k=n).
    const int rr  = n & 31;
    const int sP  = n >> 5;
    const int bhk = (rr >> 3) & 1;
    const int eP  = (rr & 7) + 8 * (rr >> 4);
    __bf16* __restrict__ of =
        outFrag + ((((long)mt * 16 + sP) * 32) << 4) + eP;
#pragma unroll
    for (int r = 0; r < 8; ++r) {
      float v = acc[r] + bn;
      v = v > 0.0f ? v : 0.0f;
      const int laneP = (r + 8 * bh) + 16 * bhk;   // (m&15) + 16*k-half
      of[laneP << 4] = f2bf(v);
    }
  } else {
#pragma unroll
    for (int r = 0; r < 8; ++r) {
      const int row = mt * 16 + r + 8 * bh;        // batch index
      outF32[(long)row * ND_ + n] = dt[row] * tanhf(acc[r] + bn);
    }
  }
}

// ztmp_frag = frag_image( z + c * k )   (frag-only output; fp32 copy not needed)
__global__ __launch_bounds__(256)
void axpy_frag_kernel(const float* __restrict__ z, const float* __restrict__ k,
                      float c, __bf16* __restrict__ outFrag) {
  int i = blockIdx.x * blockDim.x + threadIdx.x;
  if (i >= B_ * H_) return;
  int m = i >> 9, kk = i & 511;
  outFrag[frag_idx(m, kk)] = f2bf(z[i] + c * k[i]);
}

// z += c*(k1 + 2k2 + 2k3 + k4); also refresh z's fragment image
__global__ __launch_bounds__(256)
void rk4_combine_kernel(float* __restrict__ z, const float* __restrict__ k1,
                        const float* __restrict__ k2, const float* __restrict__ k3,
                        const float* __restrict__ k4, float c,
                        __bf16* __restrict__ zfrag) {
  int i = blockIdx.x * blockDim.x + threadIdx.x;
  if (i >= B_ * H_) return;
  float v = z[i] + c * (k1[i] + 2.0f * k2[i] + 2.0f * k3[i] + k4[i]);
  z[i] = v;
  int m = i >> 9, kk = i & 511;
  zfrag[frag_idx(m, kk)] = f2bf(v);
}

// poses[b, j, :] = leaky_relu(z[b] @ Wr1 + br1, 0.1) @ Wr2 + br2
__global__ __launch_bounds__(128)
void regressor_kernel(const float* __restrict__ z, const float* __restrict__ Wr1,
                      const float* __restrict__ br1, const float* __restrict__ Wr2,
                      const float* __restrict__ br2, float* __restrict__ poses,
                      int j) {
  __shared__ float hr[128];
  const int b = blockIdx.x, t = threadIdx.x;
  const float* __restrict__ zb = z + (long)b * H_;
  float a = br1[t];
  for (int k = 0; k < H_; ++k) a += zb[k] * Wr1[(long)k * 128 + t];
  hr[t] = a > 0.0f ? a : 0.1f * a;
  __syncthreads();
  if (t < 6) {
    float o = br2[t];
#pragma unroll 8
    for (int k = 0; k < 128; ++k) o += hr[k] * Wr2[k * 6 + t];
    poses[((long)b * NEV_ + j) * 6 + t] = o;
  }
}

__global__ __launch_bounds__(256)
void copy_kernel(const float* __restrict__ src, float* __restrict__ dst) {
  int i = blockIdx.x * blockDim.x + threadIdx.x;
  if (i < B_ * H_) dst[i] = src[i];
}

// ---------------------------------------------------------------------------
extern "C" void kernel_launch(void* const* d_in, const int* in_sizes, int n_in,
                              void* d_out, int out_size, void* d_ws, size_t ws_size,
                              hipStream_t stream) {
  // inputs: fv fi ts Wf0 bf0 Wf1 bf1 Wout bout Wr1 br1 Wr2 br2
  // fv (d_in[0]) / fi (d_in[1]) mathematically cancel out of the output.
  const float* ts   = (const float*)d_in[2];
  const float* Wf0  = (const float*)d_in[3];
  const float* bf0  = (const float*)d_in[4];
  const float* Wf1  = (const float*)d_in[5];
  const float* bf1  = (const float*)d_in[6];
  const float* Wout = (const float*)d_in[7];
  const float* bout = (const float*)d_in[8];
  const float* Wr1  = (const float*)d_in[9];
  const float* br1  = (const float*)d_in[10];
  const float* Wr2  = (const float*)d_in[11];
  const float* br2  = (const float*)d_in[12];
  float* out = (float*)d_out;              // [64*10*6] poses, then [64*512] h_last

  // workspace layout (~2.4 MB); all offsets 256B-aligned
  char* ws = (char*)d_ws;
  __bf16* Wf0p  = (__bf16*)(ws + 0);        // 512 KB
  __bf16* Wf1p  = (__bf16*)(ws + 524288);   // 512 KB
  __bf16* Wselp = (__bf16*)(ws + 1048576);  // 512 KB
  float*  bsel  = (float*)(ws + 1572864);   // 2 KB
  float*  dtv   = (float*)(ws + 1574912);   // 256 B
  float*  z     = (float*)(ws + 1575168);   // 128 KB
  float*  k1    = (float*)(ws + 1706240);   // 128 KB
  float*  k2    = (float*)(ws + 1837312);   // 128 KB
  float*  k3    = (float*)(ws + 1968384);   // 128 KB
  float*  k4    = (float*)(ws + 2099456);   // 128 KB
  __bf16* zfrag = (__bf16*)(ws + 2230528);  // 64 KB
  __bf16* tfrag = (__bf16*)(ws + 2296064);  // 64 KB (ztmp frag)
  __bf16* h1f   = (__bf16*)(ws + 2361600);  // 64 KB
  __bf16* h2f   = (__bf16*)(ws + 2427136);  // 64 KB

  // one-time weight packing
  const int PKB = (H_ * KD_ + 255) / 256;
  pack_w_kernel<<<PKB, 256, 0, stream>>>(Wf0, (long)H_, 1L, Wf0p);
  pack_w_kernel<<<PKB, 256, 0, stream>>>(Wf1, (long)H_, 1L, Wf1p);
  // Wsel[k][h] = Wout[k*(H*C) + h*C]
  pack_w_kernel<<<PKB, 256, 0, stream>>>(Wout, (long)H_ * C_, (long)C_, Wselp);
  prep_misc_kernel<<<(B_ * H_ + 255) / 256, 256, 0, stream>>>(ts, bout, bsel, dtv,
                                                              z, zfrag);

  // pose at z0
  regressor_kernel<<<B_, 128, 0, stream>>>(z, Wr1, br1, Wr2, br2, out, 0);

  const dim3 ggrid(4, 32);
  const int EB = (B_ * H_ + 255) / 256;
  auto feval = [&](const __bf16* zinFrag, float* kout) {
    gemm_wmma_kernel<<<ggrid, 32, 0, stream>>>(zinFrag, Wf0p,  bf0,  nullptr,
                                               h1f, nullptr, 0);
    gemm_wmma_kernel<<<ggrid, 32, 0, stream>>>(h1f,     Wf1p,  bf1,  nullptr,
                                               h2f, nullptr, 0);
    gemm_wmma_kernel<<<ggrid, 32, 0, stream>>>(h2f,     Wselp, bsel, dtv,
                                               nullptr, kout, 1);
  };

  const float h = 0.1f;
  for (int stp = 0; stp < 9; ++stp) {
    feval(zfrag, k1);
    axpy_frag_kernel<<<EB, 256, 0, stream>>>(z, k1, 0.5f * h, tfrag);
    feval(tfrag, k2);
    axpy_frag_kernel<<<EB, 256, 0, stream>>>(z, k2, 0.5f * h, tfrag);
    feval(tfrag, k3);
    axpy_frag_kernel<<<EB, 256, 0, stream>>>(z, k3, h, tfrag);
    feval(tfrag, k4);
    rk4_combine_kernel<<<EB, 256, 0, stream>>>(z, k1, k2, k3, k4, h / 6.0f, zfrag);
    regressor_kernel<<<B_, 128, 0, stream>>>(z, Wr1, br1, Wr2, br2, out, stp + 1);
  }

  // second output: h_i[:, -1] = final z
  copy_kernel<<<EB, 256, 0, stream>>>(z, out + (long)B_ * NEV_ * 6);
}
